// Global_Critic_23862838297449
// MI455X (gfx1250) — matrix operand order repaired and checked by reading, not compiled, not run on hardware
//
#include <hip/hip_runtime.h>

// ---------------------------------------------------------------------------
// Problem constants (from reference)
// ---------------------------------------------------------------------------
#define NA   8        // agents
#define NB   65536    // batch
#define OBS  128
#define ACT  32
#define KX   160      // OBS+ACT
#define E    128      // embedding dim
#define HD   128      // H*DH
#define F1K  256      // E + H*DH

#define BT   32       // batch tile per workgroup
#define SRX  168      // LDS row stride (bf16) for x   (K=160 padded)
#define SRE  136      // LDS row stride (bf16) for e/v (N=128 padded)

typedef __attribute__((ext_vector_type(16))) __bf16 bf16x16;
typedef __attribute__((ext_vector_type(8)))  __bf16 bf16x8;
typedef __attribute__((ext_vector_type(4)))  __bf16 bf16x4;
typedef __attribute__((ext_vector_type(8)))  float  v8f;
typedef __attribute__((ext_vector_type(4)))  float  v4f;

__device__ __forceinline__ __bf16 f2bf(float f)  { return (__bf16)f; }
__device__ __forceinline__ float  bf2f(__bf16 b) { return (float)b;  }
__device__ __forceinline__ float  lrelu(float x) { return x > 0.f ? x : 0.01f * x; }
__device__ __forceinline__ v8f zero8() {
    v8f z;
#pragma unroll
    for (int i = 0; i < 8; ++i) z[i] = 0.f;
    return z;
}

// A-fragment: 16x32 bf16 from row-major bf16 buffer (LDS).
// Layout used here: lanes 0-15 -> M=0..15, K=k0+half*16 .. +16 contiguous.
__device__ __forceinline__ bf16x16 loadA(const __bf16* base, int stride,
                                         int m0, int k0, int ln, int half) {
    const __bf16* p = base + (m0 + ln) * stride + k0 + half * 16;
    bf16x8 lo = *(const bf16x8*)p;
    bf16x8 hi = *(const bf16x8*)(p + 8);
    return __builtin_shufflevector(lo, hi, 0,1,2,3,4,5,6,7,8,9,10,11,12,13,14,15);
}

// B-fragment: 32x16 bf16 from N-major weight (W[N][K], K contiguous).
// Layout used here (same k order as loadA): lanes 0-15 hold K=k0..k0+15 for N=n, lanes 16-31 hold K=k0+16..k0+31.
__device__ __forceinline__ bf16x16 loadB(const __bf16* base, int strideK,
                                         int k0, int n, int half) {
    return *(const bf16x16*)(base + n * strideK + k0 + half * 16);
}

#define WMMA_BF16(Cacc, Afrag, Bfrag) \
    __builtin_amdgcn_wmma_f32_16x16x32_bf16(false, (Afrag), false, (Bfrag), \
                                            (short)0, (Cacc), false, false)

// ---------------------------------------------------------------------------
// Kernel 0: transpose + fp32->bf16 weight prep into workspace.
//   WgT  : [A][N=128][K=160]   Wf1T : [A][N=128][K=256]   WvT : [N=128][K=128]
// ---------------------------------------------------------------------------
__global__ __launch_bounds__(256) void prep_kernel(
    const float* __restrict__ Wg, const float* __restrict__ Wv,
    const float* __restrict__ Wf1,
    __bf16* __restrict__ WgT, __bf16* __restrict__ WvT, __bf16* __restrict__ Wf1T)
{
    int i = blockIdx.x * 256 + threadIdx.x;
    if (i < NA * E * KX) {                     // WgT[a][n][k] = Wg[a][k][n]
        int a = i / (E * KX); int r = i % (E * KX);
        int n = r / KX;       int k = r % KX;
        WgT[i] = f2bf(Wg[(a * KX + k) * E + n]);
        return;
    }
    i -= NA * E * KX;
    if (i < HD * E) {                          // WvT[c=h*32+d][e] = Wv[h][e][d]
        int c = i >> 7; int e = i & 127;
        int h = c >> 5; int d = c & 31;
        WvT[i] = f2bf(Wv[((h * E) + e) * 32 + d]);
        return;
    }
    i -= HD * E;
    if (i < NA * E * F1K) {                    // Wf1T[a][n][k] = Wf1[a][k][n]
        int a = i / (E * F1K); int r = i % (E * F1K);
        int n = r >> 8;        int k = r & 255;
        Wf1T[i] = f2bf(Wf1[(a * F1K + k) * E + n]);
    }
}

// ---------------------------------------------------------------------------
// Kernel 1: fully fused critic. One 256-thread WG (8 waves) per 32-batch tile.
// ---------------------------------------------------------------------------
__global__ __launch_bounds__(256) void critic_kernel(
    const float* __restrict__ obs, const float* __restrict__ act,
    const float* __restrict__ bg,  const float* __restrict__ bv,
    const float* __restrict__ bf1, const float* __restrict__ Wf2,
    const float* __restrict__ bf2,
    const __bf16* __restrict__ WgT, const __bf16* __restrict__ WvT,
    const __bf16* __restrict__ Wf1T, float* __restrict__ out)
{
    // LDS: x stage (10752) | e[8] (69632) | v[8] (69632) | vsum (8704) | q (128)
    __shared__ __align__(32) unsigned char smem[158848];
    __bf16* sX  = (__bf16*)(smem);
    __bf16* sE  = (__bf16*)(smem + 10752);
    __bf16* sV  = (__bf16*)(smem + 80384);
    __bf16* sVs = (__bf16*)(smem + 150016);
    float*  sQ  = (float*) (smem + 158720);
    __bf16* sXi = sX;                     // pass-2 alias (x stage is dead then)

    const int tid  = threadIdx.x;
    const int lane = tid & 31;
    const int w    = tid >> 5;            // wave id 0..7
    const int half = lane >> 4;
    const int ln   = lane & 15;
    const int mt   = w & 1;               // M-tile (2 x 16 rows)
    const int ntp  = w >> 1;              // N-tile pair (covers n-tiles 2*ntp, 2*ntp+1)
    const int n0   = ntp * 32 + ln;       // this lane's column, tile 0
    const int n1   = n0 + 16;             //                      tile 1
    const int m0   = mt * 16;
    const int b0   = blockIdx.x * BT;

    v8f vs0 = zero8(), vs1 = zero8();     // vsum accumulators (register-resident)

    // ------------------------ pass 1: e and v per agent ------------------------
    for (int a = 0; a < NA; ++a) {
        // stage x = [obs | act] tile as bf16 (float4 loads, bf16x4 LDS stores)
        // 32 rows x 40 float4-groups: groups 0..31 from obs, 32..39 from act.
        for (int i = tid; i < BT * 40; i += 256) {
            int m = i / 40, g = i % 40;
            size_t row = (size_t)(a * NB + b0 + m);
            v4f x = (g < 32) ? *(const v4f*)(obs + row * OBS + g * 4)
                             : *(const v4f*)(act + row * ACT + (g - 32) * 4);
            bf16x4 y;
#pragma unroll
            for (int j = 0; j < 4; ++j) y[j] = f2bf(x[j]);
            *(bf16x4*)(sX + m * SRX + g * 4) = y;
        }
        // overlap: prefetch next agent's x rows into cache while we do GEMMs
        if (a + 1 < NA) {
            size_t row = (size_t)((a + 1) * NB + b0 + lane);
            __builtin_prefetch(obs + row * OBS, 0, 1);
            __builtin_prefetch(act + row * ACT, 0, 1);
        }
        __syncthreads();

        // GEMM1: e = lrelu(x @ Wg + bg)   [32,160] x [160,128]
        v8f c0 = zero8(), c1 = zero8();
        const __bf16* wg = WgT + a * (KX * E);
#pragma unroll
        for (int k0 = 0; k0 < KX; k0 += 32) {
            bf16x16 af = loadA(sX, SRX, m0, k0, ln, half);
            bf16x16 b0f = loadB(wg, KX, k0, n0, half);
            bf16x16 b1f = loadB(wg, KX, k0, n1, half);
            c0 = WMMA_BF16(c0, af, b0f);
            c1 = WMMA_BF16(c1, af, b1f);
        }
        __bf16* se = sE + a * (BT * SRE);
        {
            float bb0 = bg[a * E + n0], bb1 = bg[a * E + n1];
#pragma unroll
            for (int r = 0; r < 8; ++r) {
                int m = m0 + r + 8 * half;
                se[m * SRE + n0] = f2bf(lrelu(c0[r] + bb0));
                se[m * SRE + n1] = f2bf(lrelu(c1[r] + bb1));
            }
        }
        __syncthreads();

        // GEMM2: v = lrelu(e @ Wv + bv)   [32,128] x [128,128]
        c0 = zero8(); c1 = zero8();
#pragma unroll
        for (int k0 = 0; k0 < E; k0 += 32) {
            bf16x16 af = loadA(se, SRE, m0, k0, ln, half);
            bf16x16 b0f = loadB(WvT, E, k0, n0, half);
            bf16x16 b1f = loadB(WvT, E, k0, n1, half);
            c0 = WMMA_BF16(c0, af, b0f);
            c1 = WMMA_BF16(c1, af, b1f);
        }
        __bf16* sv = sV + a * (BT * SRE);
        {
            float bb0 = bv[n0], bb1 = bv[n1];
#pragma unroll
            for (int r = 0; r < 8; ++r) {
                int m = m0 + r + 8 * half;
                float v0 = lrelu(c0[r] + bb0);
                float v1 = lrelu(c1[r] + bb1);
                sv[m * SRE + n0] = f2bf(v0);
                sv[m * SRE + n1] = f2bf(v1);
                vs0[r] += v0;                 // vsum over agents, in registers
                vs1[r] += v1;
            }
        }
    }
    // spill vsum to LDS (each wave owns disjoint tiles)
#pragma unroll
    for (int r = 0; r < 8; ++r) {
        int m = m0 + r + 8 * half;
        sVs[m * SRE + n0] = f2bf(vs0[r]);
        sVs[m * SRE + n1] = f2bf(vs1[r]);
    }
    __syncthreads();

    // ------------------ pass 2: f-MLP per agent (alpha == 1.0) ------------------
    for (int a = 0; a < NA; ++a) {
        // xi = vsum - v[a]  (the "attend to everyone but self" tensor), 4-wide
        const __bf16* sv = sV + a * (BT * SRE);
        for (int i = tid; i < BT * (HD / 4); i += 256) {
            int m = i >> 5, c = (i & 31) * 4;
            bf16x4 s = *(const bf16x4*)(sVs + m * SRE + c);
            bf16x4 t = *(const bf16x4*)(sv  + m * SRE + c);
            bf16x4 d;
#pragma unroll
            for (int j = 0; j < 4; ++j) d[j] = f2bf(bf2f(s[j]) - bf2f(t[j]));
            *(bf16x4*)(sXi + m * SRE + c) = d;
        }
        if (tid < BT) sQ[tid] = 0.f;
        __syncthreads();

        // GEMM3: h1 = lrelu([xi | e] @ Wf1 + bf1)   [32,256] x [256,128]
        v8f c0 = zero8(), c1 = zero8();
        const __bf16* wf = Wf1T + a * (E * F1K);
        const __bf16* se = sE + a * (BT * SRE);
#pragma unroll
        for (int k0 = 0; k0 < F1K; k0 += 32) {
            bf16x16 af = (k0 < HD) ? loadA(sXi, SRE, m0, k0, ln, half)
                                   : loadA(se,  SRE, m0, k0 - HD, ln, half);
            bf16x16 b0f = loadB(wf, F1K, k0, n0, half);
            bf16x16 b1f = loadB(wf, F1K, k0, n1, half);
            c0 = WMMA_BF16(c0, af, b0f);
            c1 = WMMA_BF16(c1, af, b1f);
        }
        // qv = h1 . Wf2 + bf2 : per-lane multiply, 16-lane shfl reduce, LDS atomics
        {
            float bb0 = bf1[a * E + n0], bb1 = bf1[a * E + n1];
            float w0  = Wf2[a * E + n0], w1  = Wf2[a * E + n1];
#pragma unroll
            for (int r = 0; r < 8; ++r) {
                float p = lrelu(c0[r] + bb0) * w0 + lrelu(c1[r] + bb1) * w1;
                p += __shfl_xor(p, 1, 16);
                p += __shfl_xor(p, 2, 16);
                p += __shfl_xor(p, 4, 16);
                p += __shfl_xor(p, 8, 16);
                if (ln == 0) atomicAdd(&sQ[m0 + r + 8 * half], p);
            }
        }
        __syncthreads();
        if (tid < BT) out[(size_t)a * NB + b0 + tid] = sQ[tid] + bf2[a];
        __syncthreads();   // protect sXi/sQ before next agent overwrites them
    }
}

// ---------------------------------------------------------------------------
// Launch
// ---------------------------------------------------------------------------
extern "C" void kernel_launch(void* const* d_in, const int* in_sizes, int n_in,
                              void* d_out, int out_size, void* d_ws, size_t ws_size,
                              hipStream_t stream) {
    const float* obs = (const float*)d_in[0];
    const float* act = (const float*)d_in[1];
    const float* Wg  = (const float*)d_in[2];
    const float* bg  = (const float*)d_in[3];
    // d_in[4] = Wq, d_in[5] = Wk : dead code (softmax over size-1 axis == 1.0)
    const float* Wv  = (const float*)d_in[6];
    const float* bv  = (const float*)d_in[7];
    const float* Wf1 = (const float*)d_in[8];
    const float* bf1 = (const float*)d_in[9];
    const float* Wf2 = (const float*)d_in[10];
    const float* bf2 = (const float*)d_in[11];
    float* out = (float*)d_out;

    char* ws = (char*)d_ws;
    __bf16* WgT  = (__bf16*)(ws);                    // 8*128*160*2 = 327680 B
    __bf16* WvT  = (__bf16*)(ws + 327680);           // 128*128*2   =  32768 B
    __bf16* Wf1T = (__bf16*)(ws + 360448);           // 8*128*256*2 = 524288 B

    const int prepN = NA * E * KX + HD * E + NA * E * F1K;   // 442368
    prep_kernel<<<(prepN + 255) / 256, 256, 0, stream>>>(Wg, Wv, Wf1, WgT, WvT, Wf1T);

    critic_kernel<<<NB / BT, 256, 0, stream>>>(obs, act, bg, bv, bf1, Wf2, bf2,
                                               WgT, WvT, Wf1T, out);
}
